// Graphsage_51084341018870
// MI455X (gfx1250) — compile-verified, matrix operand order checked by reading, not run on hardware
//
#include <hip/hip_runtime.h>
#include <hip/hip_bf16.h>

typedef __attribute__((ext_vector_type(16))) _Float16 v16h;
typedef __attribute__((ext_vector_type(8)))  _Float16 v8h;
typedef __attribute__((ext_vector_type(8)))  float    v8f;

static constexpr int  kN1   = 100000;
static constexpr int  kN2   = 20000;
static constexpr long kE1   = 1600000;
static constexpr long kE2   = 320000;
static constexpr int  kDIN  = 128;
static constexpr int  kDH   = 256;
static constexpr int  kDOUT = 128;

// ---------------------------------------------------------------------------
// utility kernels
// ---------------------------------------------------------------------------
__global__ void zero_f32(float* __restrict__ p, long n) {
  long i = (long)blockIdx.x * blockDim.x + threadIdx.x;
  if (i < n) p[i] = 0.0f;
}

__global__ void conv_f32_to_f16(const float* __restrict__ src,
                                _Float16* __restrict__ dst, long n) {
  long i = (long)blockIdx.x * blockDim.x + threadIdx.x;
  if (i < n) dst[i] = (_Float16)src[i];
}

// build combined weight [DHout][K0+K1] in f16 from two row-major f32 mats
__global__ void build_wcat(const float* __restrict__ Wa,   // [rows][K0]
                           const float* __restrict__ Wb,   // [rows][K1]
                           _Float16* __restrict__ Wc,      // [rows][K0+K1]
                           int rows, int K0, int K1) {
  long i = (long)blockIdx.x * blockDim.x + threadIdx.x;
  long n = (long)rows * (K0 + K1);
  if (i >= n) return;
  int r = (int)(i / (K0 + K1));
  int k = (int)(i % (K0 + K1));
  float v = (k < K0) ? Wa[(size_t)r * K0 + k] : Wb[(size_t)r * K1 + (k - K0)];
  Wc[i] = (_Float16)v;
}

// ---------------------------------------------------------------------------
// mean-aggregation: scatter-add (one wave per edge) + finalize
// ---------------------------------------------------------------------------
__global__ void scatter_f32(const float* __restrict__ feat, // [*, 128] f32
                            const int* __restrict__ src, const int* __restrict__ dst,
                            float* __restrict__ acc,        // [N1, 128]
                            float* __restrict__ cnt, long E) {
  long gid = (long)blockIdx.x * blockDim.x + threadIdx.x;
  long e = gid >> 5;
  int lane = (int)(gid & 31);
  if (e >= E) return;
  int s = src[e], d = dst[e];
  const float* xs = feat + (size_t)s * kDIN;
  float* hd = acc + (size_t)d * kDIN;
#pragma unroll
  for (int j = 0; j < kDIN / 32; ++j) {
    int f = lane + 32 * j;
    atomicAdd(&hd[f], xs[f]);
  }
  if (lane == 0) atomicAdd(&cnt[d], 1.0f);
}

__global__ void scatter_f16(const _Float16* __restrict__ feat, // [*, 256] f16
                            const int* __restrict__ src, const int* __restrict__ dst,
                            float* __restrict__ acc,           // [N2, 256]
                            float* __restrict__ cnt, long E) {
  long gid = (long)blockIdx.x * blockDim.x + threadIdx.x;
  long e = gid >> 5;
  int lane = (int)(gid & 31);
  if (e >= E) return;
  int s = src[e], d = dst[e];
  const _Float16* xs = feat + (size_t)s * kDH;
  float* hd = acc + (size_t)d * kDH;
#pragma unroll
  for (int j = 0; j < kDH / 32; ++j) {
    int f = lane + 32 * j;
    atomicAdd(&hd[f], (float)xs[f]);
  }
  if (lane == 0) atomicAdd(&cnt[d], 1.0f);
}

__global__ void finalize_mean(const float* __restrict__ acc,
                              const float* __restrict__ cnt,
                              _Float16* __restrict__ out16,
                              long n, int dim_log2) {
  long i = (long)blockIdx.x * blockDim.x + threadIdx.x;
  if (i >= n) return;
  long row = i >> dim_log2;
  float c = fmaxf(cnt[row], 1.0f);
  out16[i] = (_Float16)(acc[i] / c);
}

// ---------------------------------------------------------------------------
// WMMA fragment loader: 16-bit A/B 16x32 layout (ISA 7.12.2)
// lane L<16  : row L,  halves = K[0..7],  K[16..23]
// lane L>=16 : row L-16, halves = K[8..15], K[24..31]
// ---------------------------------------------------------------------------
static __device__ __forceinline__ v16h load_frag(const _Float16* __restrict__ p0,
                                                 int ld, int lane) {
  const int l = lane & 15;
  const int hiofs = (lane & 16) ? 8 : 0;
  const _Float16* p = p0 + (size_t)l * ld + hiofs;
  v8h lo = *(const v8h*)(p);
  v8h hi = *(const v8h*)(p + 16);
  v16h r;
#pragma unroll
  for (int i = 0; i < 8; ++i) { r[i] = lo[i]; r[i + 8] = hi[i]; }
  return r;
}

// ---------------------------------------------------------------------------
// Layer-1 fused GEMM: h = [x | h_n] @ Wc1^T + b ; L2-norm row ; BN ; ReLU
// block = 256 threads (8 waves), 16 rows x 256 cols; wave owns 2 col-tiles.
// ---------------------------------------------------------------------------
__global__ __launch_bounds__(256) void gemm1_fused(
    const _Float16* __restrict__ x16,   // [N1,128]
    const _Float16* __restrict__ hn16,  // [N1,128]
    const _Float16* __restrict__ wc1,   // [256,256] rows = out-ch, cols = k
    const float* __restrict__ bias,     // [256]
    const float* __restrict__ gamma, const float* __restrict__ beta,
    const float* __restrict__ mean, const float* __restrict__ var,
    _Float16* __restrict__ h16)         // [N1,256]
{
  const int tid = threadIdx.x;
  const int lane = tid & 31;
  const int wave = tid >> 5;
  const int m0 = blockIdx.x * 16;
  const int n0 = wave * 32;

  v8f acc0 = {}; v8f acc1 = {};
#pragma unroll
  for (int ks = 0; ks < 8; ++ks) {
    const int k = ks * 32;
    const _Float16* asrc = (k < 128)
        ? (x16 + (size_t)m0 * 128 + k)
        : (hn16 + (size_t)m0 * 128 + (k - 128));
    v16h a  = load_frag(asrc, 128, lane);
    v16h b0 = load_frag(wc1 + (size_t)n0 * 256 + k, 256, lane);
    v16h b1 = load_frag(wc1 + (size_t)(n0 + 16) * 256 + k, 256, lane);
    acc0 = __builtin_amdgcn_wmma_f32_16x16x32_f16(false, a, false, b0,
                                                  (short)0, acc0, false, false);
    acc1 = __builtin_amdgcn_wmma_f32_16x16x32_f16(false, a, false, b1,
                                                  (short)0, acc1, false, false);
  }

  __shared__ float tile[16][257];
  __shared__ float rn[16];
  const int nlo = lane & 15;
  const int mofs = (lane & 16) ? 8 : 0;
#pragma unroll
  for (int r = 0; r < 8; ++r) {
    tile[mofs + r][n0 + nlo]      = acc0[r] + bias[n0 + nlo];
    tile[mofs + r][n0 + 16 + nlo] = acc1[r] + bias[n0 + 16 + nlo];
  }
  __syncthreads();

  if (tid < 16) {
    float s = 0.0f;
    for (int c = 0; c < 256; ++c) { float v = tile[tid][c]; s += v * v; }
    rn[tid] = 1.0f / fmaxf(sqrtf(s), 1e-12f);
  }
  __syncthreads();

  const int col = tid;  // 0..255
  const float g  = gamma[col];
  const float bb = beta[col];
  const float mu = mean[col];
  const float iv = rsqrtf(var[col] + 1e-5f);
#pragma unroll 4
  for (int r = 0; r < 16; ++r) {
    float v = tile[r][col] * rn[r];
    v = g * (v - mu) * iv + bb;
    v = fmaxf(v, 0.0f);
    h16[(size_t)(m0 + r) * 256 + col] = (_Float16)v;
  }
}

// ---------------------------------------------------------------------------
// Layer-2 fused GEMM: out = [h | h_n2] @ Wc2^T + b ; L2-norm row (f32 out)
// block = 256 threads (8 waves), 16 rows x 128 cols; wave owns 1 col-tile.
// ---------------------------------------------------------------------------
__global__ __launch_bounds__(256) void gemm2_fused(
    const _Float16* __restrict__ h16,    // [N1,256] (rows 0..N2 used)
    const _Float16* __restrict__ hn2_16, // [N2,256]
    const _Float16* __restrict__ wc2,    // [128,512]
    const float* __restrict__ bias,      // [128]
    float* __restrict__ out)             // [N2,128]
{
  const int tid = threadIdx.x;
  const int lane = tid & 31;
  const int wave = tid >> 5;
  const int m0 = blockIdx.x * 16;
  const int n0 = wave * 16;

  v8f acc = {};
#pragma unroll
  for (int ks = 0; ks < 16; ++ks) {
    const int k = ks * 32;
    const _Float16* asrc = (k < 256)
        ? (h16 + (size_t)m0 * 256 + k)
        : (hn2_16 + (size_t)m0 * 256 + (k - 256));
    v16h a = load_frag(asrc, 256, lane);
    v16h b = load_frag(wc2 + (size_t)n0 * 512 + k, 512, lane);
    acc = __builtin_amdgcn_wmma_f32_16x16x32_f16(false, a, false, b,
                                                 (short)0, acc, false, false);
  }

  __shared__ float tile[16][129];
  __shared__ float rn[16];
  const int nlo = lane & 15;
  const int mofs = (lane & 16) ? 8 : 0;
#pragma unroll
  for (int r = 0; r < 8; ++r) {
    tile[mofs + r][n0 + nlo] = acc[r] + bias[n0 + nlo];
  }
  __syncthreads();

  if (tid < 16) {
    float s = 0.0f;
    for (int c = 0; c < 128; ++c) { float v = tile[tid][c]; s += v * v; }
    rn[tid] = 1.0f / fmaxf(sqrtf(s), 1e-12f);
  }
  __syncthreads();

  if (tid < 128) {
    const int col = tid;
#pragma unroll 4
    for (int r = 0; r < 16; ++r) {
      out[(size_t)(m0 + r) * 128 + col] = tile[r][col] * rn[r];
    }
  }
}

// ---------------------------------------------------------------------------
// host launcher
// ---------------------------------------------------------------------------
extern "C" void kernel_launch(void* const* d_in, const int* in_sizes, int n_in,
                              void* d_out, int out_size, void* d_ws, size_t ws_size,
                              hipStream_t stream) {
  const float* x     = (const float*)d_in[0];
  const int*   src1  = (const int*)d_in[1];
  const int*   dst1  = (const int*)d_in[2];
  const int*   src2  = (const int*)d_in[3];
  const int*   dst2  = (const int*)d_in[4];
  const float* W1_1  = (const float*)d_in[7];
  const float* W2_1  = (const float*)d_in[8];
  const float* b2_1  = (const float*)d_in[9];
  const float* gamma = (const float*)d_in[10];
  const float* beta  = (const float*)d_in[11];
  const float* mean  = (const float*)d_in[12];
  const float* var   = (const float*)d_in[13];
  const float* W1_2  = (const float*)d_in[14];
  const float* W2_2  = (const float*)d_in[15];
  const float* b2_2  = (const float*)d_in[16];
  float* out = (float*)d_out;

  // workspace carve-out (256B aligned)
  size_t o = 0;
  auto carve = [&](size_t bytes) {
    size_t r = o;
    o = (o + bytes + 255) & ~(size_t)255;
    return r;
  };
  char* ws = (char*)d_ws;
  float*    hn1    = (float*)(ws + carve((size_t)kN1 * kDIN * 4));
  float*    cnt1   = (float*)(ws + carve((size_t)kN1 * 4));
  _Float16* x16    = (_Float16*)(ws + carve((size_t)kN1 * kDIN * 2));
  _Float16* hn16   = (_Float16*)(ws + carve((size_t)kN1 * kDIN * 2));
  _Float16* wc1    = (_Float16*)(ws + carve((size_t)kDH * (2 * kDIN) * 2));
  _Float16* h16    = (_Float16*)(ws + carve((size_t)kN1 * kDH * 2));
  float*    hn2    = (float*)(ws + carve((size_t)kN2 * kDH * 4));
  float*    cnt2   = (float*)(ws + carve((size_t)kN2 * 4));
  _Float16* hn2_16 = (_Float16*)(ws + carve((size_t)kN2 * kDH * 2));
  _Float16* wc2    = (_Float16*)(ws + carve((size_t)kDOUT * (2 * kDH) * 2));
  (void)ws_size; (void)in_sizes; (void)n_in; (void)out_size;

  const int T = 256;
  auto blocks = [&](long n) { return (unsigned)((n + T - 1) / T); };

  // 0) zero accumulators (graph-safe, deterministic)
  zero_f32<<<blocks((long)kN1 * kDIN), T, 0, stream>>>(hn1, (long)kN1 * kDIN);
  zero_f32<<<blocks(kN1), T, 0, stream>>>(cnt1, kN1);
  zero_f32<<<blocks((long)kN2 * kDH), T, 0, stream>>>(hn2, (long)kN2 * kDH);
  zero_f32<<<blocks(kN2), T, 0, stream>>>(cnt2, kN2);

  // 1) precision conversions / combined weights
  conv_f32_to_f16<<<blocks((long)kN1 * kDIN), T, 0, stream>>>(x, x16, (long)kN1 * kDIN);
  build_wcat<<<blocks((long)kDH * 2 * kDIN), T, 0, stream>>>(W1_1, W2_1, wc1, kDH, kDIN, kDIN);
  build_wcat<<<blocks((long)kDOUT * 2 * kDH), T, 0, stream>>>(W1_2, W2_2, wc2, kDOUT, kDH, kDH);

  // 2) layer-1 mean aggregation
  scatter_f32<<<blocks(kE1 * 32), T, 0, stream>>>(x, src1, dst1, hn1, cnt1, kE1);
  finalize_mean<<<blocks((long)kN1 * kDIN), T, 0, stream>>>(hn1, cnt1, hn16,
                                                            (long)kN1 * kDIN, 7);

  // 3) layer-1 fused GEMM + l2norm + BN + ReLU  (WMMA)
  gemm1_fused<<<kN1 / 16, 256, 0, stream>>>(x16, hn16, wc1, b2_1,
                                            gamma, beta, mean, var, h16);

  // 4) layer-2 mean aggregation (from f16 activations, f32 accumulate)
  scatter_f16<<<blocks(kE2 * 32), T, 0, stream>>>(h16, src2, dst2, hn2, cnt2, kE2);
  finalize_mean<<<blocks((long)kN2 * kDH), T, 0, stream>>>(hn2, cnt2, hn2_16,
                                                           (long)kN2 * kDH, 8);

  // 5) layer-2 fused GEMM + l2norm  (WMMA) -> f32 output
  gemm2_fused<<<kN2 / 16, 256, 0, stream>>>(h16, hn2_16, wc2, b2_2, out);
}
